// TopKSparseAttention_1425929142647
// MI455X (gfx1250) — matrix operand order repaired and checked by reading, not compiled, not run on hardware
//
#include <hip/hip_runtime.h>
#include <hip/hip_bf16.h>
#include <cstdint>
#include <cstddef>

// Problem constants (fixed by the reference)
#define B_    4
#define L_    1024
#define E_    1024
#define H_    16
#define TOPK_ 32
#define DH_   64
#define M_    (B_ * L_)   // 4096 rows for all GEMMs

typedef __attribute__((ext_vector_type(16))) __bf16         v16bf;
typedef __attribute__((ext_vector_type(8)))  float          v8f;
typedef __attribute__((ext_vector_type(8)))  unsigned short us8;
typedef __attribute__((ext_vector_type(4)))  unsigned int   u32x4;
typedef __attribute__((ext_vector_type(8)))  int            i32x8;
typedef __attribute__((ext_vector_type(4)))  int            i32x4;

static __device__ __forceinline__ float bf2f(unsigned short u) {
  union { unsigned int i; float f; } x; x.i = ((unsigned int)u) << 16; return x.f;
}
static __device__ __forceinline__ unsigned short f2bf(float f) {
  union { float f; unsigned int i; } x; x.f = f;
  unsigned int r = x.i + 0x7fffu + ((x.i >> 16) & 1u);   // round-to-nearest-even
  return (unsigned short)(r >> 16);
}

// B-fragment (32x16 KxN): lane n<16 = col n, K 0..15; lanes 16..31 = K 16..31.
// Per lane: 16 contiguous bf16 (works for global or LDS pointers).
static __device__ __forceinline__ v16bf ldB16(const unsigned short* p) {
  union { us8 s[2]; v16bf v; } u;
  u.s[0] = *(const us8*)(p);
  u.s[1] = *(const us8*)(p + 8);
  return u.v;
}
// A-fragment (16x32 MxK, 16-bit): 16 contiguous K kb..kb+15
// (kb = 0 lanes 0-15, 16 lanes 16-31, folded in by caller).
static __device__ __forceinline__ v16bf ldAfrag(const unsigned short* rowp, int kb) {
  union { us8 s[2]; v16bf v; } u;
  u.s[0] = *(const us8*)(rowp + kb);
  u.s[1] = *(const us8*)(rowp + kb + 8);
  return u.v;
}

static __device__ __forceinline__ void wait_tensorcnt0() {
#if __has_builtin(__builtin_amdgcn_s_wait_tensorcnt)
  __builtin_amdgcn_s_wait_tensorcnt(0);
#else
  asm volatile("s_wait_tensorcnt 0x0" ::: "memory");
#endif
}

// ---------------------------------------------------------------------------
// f32 -> bf16 conversion (activations and weights)
// ---------------------------------------------------------------------------
__global__ __launch_bounds__(256) void f32_to_bf16_kernel(
    const float* __restrict__ in, unsigned short* __restrict__ out, int n) {
  int i = (blockIdx.x * 256 + threadIdx.x) * 4;
  if (i + 3 < n) {
    float4 v = *(const float4*)(in + i);
    out[i + 0] = f2bf(v.x);
    out[i + 1] = f2bf(v.y);
    out[i + 2] = f2bf(v.z);
    out[i + 3] = f2bf(v.w);
  } else {
    for (; i < n; ++i) out[i] = f2bf(in[i]);
  }
}

// ---------------------------------------------------------------------------
// NT GEMM: C[m,n] = sum_k X[m,k] * W[n,k] + bias[n]   (X:[M,E] bf16, W:[E,E] bf16)
// Block tile: 256(M) x 64(N), 8 waves x (32x64): each wave owns TWO 16-row
// M-tiles so every LDS B-fragment feeds two WMMAs (hides ds latency, doubles
// math per s_wait_dscnt). The 64x32 W tile per K-step is shared by all waves:
// TDM (tensor_load_to_lds) streams it into double-buffered LDS while WMMAs
// consume the previous one.
// mode 0: store bf16 head-major [B,H,L,DH]; mode 1: store f32 row-major [M,E].
// ---------------------------------------------------------------------------
__global__ __launch_bounds__(256) void gemm_nt_bf16(
    const unsigned short* __restrict__ X, const unsigned short* __restrict__ W,
    const float* __restrict__ bias, unsigned short* __restrict__ outb,
    float* __restrict__ outf, int mode) {
  __shared__ __align__(128) unsigned short wtile[2][64 * 32];  // 2 x 4 KB

  const int lane  = threadIdx.x & 31;
  const int warp  = threadIdx.x >> 5;
  const int bm    = blockIdx.x >> 4;            // 0..15  (M / 256)
  const int nbase = (blockIdx.x & 15) * 64;     // 0..960 (N / 64)
  const int lo16  = (lane < 16);
  const int akb   = lo16 ? 0 : 16;              // A K-base per half-wave
  const int bkb   = lo16 ? 0 : 16;              // B K-base per half-wave
  const int mrow0 = (bm << 8) + (warp << 5) + (lane & 15);   // wave M-tile 0
  const unsigned short* xrow0 = X + (size_t)mrow0 * E_;
  const unsigned short* xrow1 = xrow0 + (size_t)16 * E_;     // wave M-tile 1

#if __has_builtin(__builtin_amdgcn_tensor_load_to_lds)
  #define TDM_PATH 1
#else
  #define TDM_PATH 0
#endif

  auto issue_wtile = [&](int buf, int k0) {
#if TDM_PATH
    if (threadIdx.x < 32) {   // one wave issues the DMA (EXEC ignored by TDM)
      const uint64_t ga = (uint64_t)(uintptr_t)W +
                          ((uint64_t)nbase * E_ + (uint64_t)k0) * 2u;
      const unsigned int ldsoff = (unsigned int)(uintptr_t)&wtile[buf][0];
      // D# group 0: count=1 | lds_addr | global_addr(57b) | type=2
      u32x4 g0;
      g0[0] = 1u;                                   // count = 1 valid descriptor
      g0[1] = ldsoff;                               // lds_addr
      g0[2] = (unsigned int)ga;                     // global_addr[31:0]
      g0[3] = (unsigned int)((ga >> 32) & 0x01FFFFFFu) | 0x80000000u; // [56:32] | type=2
      // D# group 1: data_size=2B, tensor 1024x1024, tile 32x64, stride0=1024
      i32x8 g1;
      g1[0] = (int)(1u << 16);                      // data_size = 1 (2 bytes)
      g1[1] = (int)(((unsigned)E_ & 0xFFFFu) << 16);// tensor_dim0[15:0] << 16
      g1[2] = (int)(((unsigned)E_ & 0xFFFFu) << 16);// dim0_hi=0 | tensor_dim1[15:0]
      g1[3] = (int)(32u << 16);                     // dim1_hi=0 | tile_dim0=32
      g1[4] = 64;                                   // tile_dim1=64 | tile_dim2=0
      g1[5] = E_;                                   // tensor_dim0_stride[31:0]
      g1[6] = 0;                                    // stride0_hi | stride1_lo
      g1[7] = 0;
      i32x4 z4 = {};
      i32x8 z8 = {};
      __builtin_amdgcn_tensor_load_to_lds(g0, g1, z4, z4, z8, 0);
    }
#else
    // fallback: 256 threads copy 64 rows x 64B synchronously
    const int row = threadIdx.x >> 2, chunk = threadIdx.x & 3;
    const us8 v = *(const us8*)(W + (size_t)(nbase + row) * E_ + k0 + chunk * 8);
    *(us8*)&wtile[buf][row * 32 + chunk * 8] = v;
#endif
  };

  v8f acc[8];                      // [t*4+j]: t = M subtile, j = N subtile
#pragma unroll
  for (int j = 0; j < 8; ++j) acc[j] = {};

  // prologue: buffer 0 <- K-step 0; A fragments for step 0 in registers
  issue_wtile(0, 0);
  v16bf a0_cur = ldAfrag(xrow0, 0 + akb);
  v16bf a1_cur = ldAfrag(xrow1, 0 + akb);
#if TDM_PATH
  if (threadIdx.x < 32) wait_tensorcnt0();
#endif
  __syncthreads();

  for (int k0 = 0; k0 < E_; k0 += 32) {
    const int cur = (k0 >> 5) & 1;
    const bool last = (k0 + 32 >= E_);
    // kick off next weight tile into the other buffer (overlaps with WMMAs)
    if (!last) issue_wtile(cur ^ 1, k0 + 32);
    // prefetch + load next A fragments early (register double-buffer)
    v16bf a0_nxt, a1_nxt;
    if (!last) {
      __builtin_prefetch(xrow0 + k0 + 32 + akb, 0, 3);
      a0_nxt = ldAfrag(xrow0, k0 + 32 + akb);
      a1_nxt = ldAfrag(xrow1, k0 + 32 + akb);
    }
    // Each B fragment feeds two WMMAs (two M subtiles) -> LDS latency is
    // amortized over twice the matrix work.
#pragma unroll
    for (int j = 0; j < 4; ++j) {
      v16bf b = ldB16(&wtile[cur][((j << 4) + (lane & 15)) * 32 + bkb]);
      acc[j] = __builtin_amdgcn_wmma_f32_16x16x32_bf16(
          false, a0_cur, false, b, (short)0, acc[j], false, false);
      acc[4 + j] = __builtin_amdgcn_wmma_f32_16x16x32_bf16(
          false, a1_cur, false, b, (short)0, acc[4 + j], false, false);
    }
    a0_cur = a0_nxt;
    a1_cur = a1_nxt;
#if TDM_PATH
    if (!last && threadIdx.x < 32) wait_tensorcnt0();
#endif
    __syncthreads();
  }

  // C/D layout: VGPR r, lanes 0-15 -> M=r, lanes 16-31 -> M=r+8; N = lane%16.
#pragma unroll
  for (int t = 0; t < 2; ++t) {
#pragma unroll
    for (int j = 0; j < 4; ++j) {
      const int   n  = nbase + (j << 4) + (lane & 15);
      const float bn = bias[n];
#pragma unroll
      for (int r = 0; r < 8; ++r) {
        const int   m  = (bm << 8) + (warp << 5) + (t << 4) + r + (lo16 ? 0 : 8);
        const float vv = acc[t * 4 + j][r] + bn;
        if (mode == 0) {
          const int b = m >> 10, l = m & (L_ - 1);
          const int h = n >> 6,  d = n & (DH_ - 1);
          outb[((((size_t)b * H_ + h) * L_ + l) << 6) + d] = f2bf(vv);
        } else {
          outf[(size_t)m * E_ + n] = vv;
        }
      }
    }
  }
}

// ---------------------------------------------------------------------------
// Top-k sparse attention: one wave per 16-query tile of one (b,h).
// logits via WMMA (key fragments register-double-buffered) -> LDS transpose ->
// register-resident exact top-32 -> softmax over 32 -> sparse V gather.
// ---------------------------------------------------------------------------
__global__ __launch_bounds__(32) void attn_topk(
    const unsigned short* __restrict__ qh, const unsigned short* __restrict__ kh,
    const unsigned short* __restrict__ vh, unsigned short* __restrict__ gctx) {
  extern __shared__ float lds[];                 // 16 * 1024 f32 = 64 KB
  const int lane  = threadIdx.x & 31;
  const int qtile = blockIdx.x & (L_ / 16 - 1);  // 0..63
  const int bh    = blockIdx.x >> 6;             // b*H + h
  const int b     = bh >> 4, h = bh & (H_ - 1);
  const size_t headoff = (size_t)bh * L_ * DH_;
  const int lo16 = (lane < 16);
  const int akb  = lo16 ? 0 : 16;
  const int bkb  = lo16 ? 0 : 16;

  const int qrow = (qtile << 4) + (lane & 15);
  const unsigned short* qp = qh + headoff + (size_t)qrow * DH_;
  const v16bf qa0 = ldAfrag(qp, 0 + akb);
  const v16bf qa1 = ldAfrag(qp, 32 + akb);
  const float scale = 0.125f;                    // 1/sqrt(64)

  const unsigned short* kbase = kh + headoff + (size_t)(lane & 15) * DH_;
  // register double-buffer of the key-tile B fragments
  v16bf kb0 = ldB16(kbase + 0 + bkb);
  v16bf kb1 = ldB16(kbase + 32 + bkb);

  for (int nt = 0; nt < L_ / 16; ++nt) {
    v16bf kn0, kn1;
    if (nt + 1 < L_ / 16) {                      // issue next tile's loads early
      const unsigned short* kp = kbase + (size_t)((nt + 1) << 4) * DH_;
      kn0 = ldB16(kp + 0 + bkb);
      kn1 = ldB16(kp + 32 + bkb);
    }
    v8f acc = {};
    acc = __builtin_amdgcn_wmma_f32_16x16x32_bf16(false, qa0, false, kb0, (short)0, acc, false, false);
    acc = __builtin_amdgcn_wmma_f32_16x16x32_bf16(false, qa1, false, kb1, (short)0, acc, false, false);
    const int col = (nt << 4) + (lane & 15);
#pragma unroll
    for (int r = 0; r < 8; ++r) {
      const int row = r + (lo16 ? 0 : 8);
      lds[row * L_ + col] = acc[r] * scale;
    }
    kb0 = kn0; kb1 = kn1;
  }
  __syncthreads();  // single-wave WG: S_NOP, but orders LDS for the compiler

  for (int r = 0; r < 16; ++r) {
    float cv[32];
#pragma unroll
    for (int j = 0; j < 32; ++j) cv[j] = lds[r * L_ + (j << 5) + lane];

    float lmv = -INFINITY; int lmi = lane;
#pragma unroll
    for (int j = 0; j < 32; ++j) {
      float c = cv[j]; int gi = (j << 5) | lane;
      if (c > lmv) { lmv = c; lmi = gi; }
    }

    float selv = 0.f; int seli = 0;                // lane t holds t-th pick
    for (int t = 0; t < TOPK_; ++t) {
      float wv = lmv; int wi = lmi;
#pragma unroll
      for (int off = 16; off >= 1; off >>= 1) {    // wave argmax
        float ov = __shfl_xor(wv, off, 32);
        int   oi = __shfl_xor(wi, off, 32);
        if (ov > wv || (ov == wv && oi < wi)) { wv = ov; wi = oi; }
      }
      if (lane == t) { selv = wv; seli = wi; }
      if ((wi & 31) == lane) {                     // owner invalidates + rescans
        const int jj = wi >> 5;
        lmv = -INFINITY; lmi = lane;
#pragma unroll
        for (int j = 0; j < 32; ++j) {
          if (j == jj) cv[j] = -INFINITY;
          float c = cv[j]; int gi = (j << 5) | lane;
          if (c > lmv) { lmv = c; lmi = gi; }
        }
      }
    }

    const float mx = __shfl(selv, 0, 32);          // pick 0 is the row max
    float e = __expf(selv - mx);
    float s = e;
#pragma unroll
    for (int off = 16; off >= 1; off >>= 1) s += __shfl_xor(s, off, 32);
    const float p = e / s;

    float a0 = 0.f, a1 = 0.f;                      // sparse attn @ V
    for (int t = 0; t < TOPK_; ++t) {
      const float pt = __shfl(p, t, 32);
      const int   it = __shfl(seli, t, 32);
      const unsigned short* vr = vh + headoff + (size_t)it * DH_;
      a0 += pt * bf2f(vr[lane]);
      a1 += pt * bf2f(vr[lane + 32]);
    }

    const int l = (qtile << 4) + r;
    unsigned short* op = gctx + ((size_t)b * L_ + l) * E_ + h * DH_;
    op[lane]      = f2bf(a0);
    op[lane + 32] = f2bf(a1);
  }
}

// ---------------------------------------------------------------------------
// Host launcher
// ---------------------------------------------------------------------------
extern "C" void kernel_launch(void* const* d_in, const int* in_sizes, int n_in,
                              void* d_out, int out_size, void* d_ws, size_t ws_size,
                              hipStream_t stream) {
  const float* Q  = (const float*)d_in[0];
  const float* K  = (const float*)d_in[1];
  const float* V  = (const float*)d_in[2];
  const float* Wq = (const float*)d_in[3];
  const float* bq = (const float*)d_in[4];
  const float* Wk = (const float*)d_in[5];
  const float* bk = (const float*)d_in[6];
  const float* Wv = (const float*)d_in[7];
  const float* bv = (const float*)d_in[8];
  const float* Wo = (const float*)d_in[9];
  const float* bo = (const float*)d_in[10];
  float* out = (float*)d_out;

  char* ws = (char*)d_ws;
  const size_t SZ_ACT = (size_t)M_ * E_ * sizeof(unsigned short);  // 8 MB
  const size_t SZ_W   = (size_t)E_ * E_ * sizeof(unsigned short);  // 2 MB
  unsigned short* Qb  = (unsigned short*)(ws + 0 * SZ_ACT);
  unsigned short* Kb  = (unsigned short*)(ws + 1 * SZ_ACT);
  unsigned short* Vb  = (unsigned short*)(ws + 2 * SZ_ACT);
  unsigned short* Wqb = (unsigned short*)(ws + 3 * SZ_ACT + 0 * SZ_W);
  unsigned short* Wkb = (unsigned short*)(ws + 3 * SZ_ACT + 1 * SZ_W);
  unsigned short* Wvb = (unsigned short*)(ws + 3 * SZ_ACT + 2 * SZ_W);
  unsigned short* Wob = (unsigned short*)(ws + 3 * SZ_ACT + 3 * SZ_W);
  unsigned short* qhb = (unsigned short*)(ws + 3 * SZ_ACT + 4 * SZ_W);
  unsigned short* khb = (unsigned short*)(ws + 4 * SZ_ACT + 4 * SZ_W);
  unsigned short* vhb = (unsigned short*)(ws + 5 * SZ_ACT + 4 * SZ_W);
  unsigned short* ctx = (unsigned short*)(ws + 6 * SZ_ACT + 4 * SZ_W);
  // total workspace use: 7*8MB + 4*2MB = 64 MB

  const int NA = M_ * E_;  // 4M elements
  const int NW = E_ * E_;  // 1M elements
  f32_to_bf16_kernel<<<NA / 1024, 256, 0, stream>>>(Q,  Qb,  NA);
  f32_to_bf16_kernel<<<NA / 1024, 256, 0, stream>>>(K,  Kb,  NA);
  f32_to_bf16_kernel<<<NA / 1024, 256, 0, stream>>>(V,  Vb,  NA);
  f32_to_bf16_kernel<<<NW / 1024, 256, 0, stream>>>(Wq, Wqb, NW);
  f32_to_bf16_kernel<<<NW / 1024, 256, 0, stream>>>(Wk, Wkb, NW);
  f32_to_bf16_kernel<<<NW / 1024, 256, 0, stream>>>(Wv, Wvb, NW);
  f32_to_bf16_kernel<<<NW / 1024, 256, 0, stream>>>(Wo, Wob, NW);

  const int gblocks = (M_ / 256) * (E_ / 64);  // 16 * 16 = 256 blocks
  gemm_nt_bf16<<<gblocks, 256, 0, stream>>>(Qb, Wqb, bq, qhb, nullptr, 0);
  gemm_nt_bf16<<<gblocks, 256, 0, stream>>>(Kb, Wkb, bk, khb, nullptr, 0);
  gemm_nt_bf16<<<gblocks, 256, 0, stream>>>(Vb, Wvb, bv, vhb, nullptr, 0);

  attn_topk<<<B_ * H_ * (L_ / 16), 32, 16 * L_ * sizeof(float), stream>>>(
      qhb, khb, vhb, ctx);

  gemm_nt_bf16<<<gblocks, 256, 0, stream>>>(ctx, Wob, bo, nullptr, out, 1);
}
